// RowAttentionWithPairBias_2199023255606
// MI455X (gfx1250) — compile-verified, matrix operand order checked by reading, not run on hardware
//
#include <hip/hip_runtime.h>
#include <hip/hip_bf16.h>
#include <math.h>

// ---------------------------------------------------------------------------
// RowAttentionWithPairBias for MI455X (gfx1250), wave32 + WMMA bf16.
// Shapes: N=384 res, A=640 atoms, M=4 conformers, C1=256, C2=128, AC=32, H=8.
// Rows buffer layout: rows 0..383 = rec, rows 384..2943 = lig (m*640+a).
// Channel layout for Q/K/V/att: permuted to [h*32+c] (head-contiguous);
// reference layout is [c*8+h] -> permutation folded into packed weights.
// All weights pre-packed to bf16, transposed so B-operand loads are
// contiguous 16B runs per lane (global_load_b128 instead of 16x b32+cvt).
// ---------------------------------------------------------------------------

typedef __bf16 bf16_t;
typedef __attribute__((ext_vector_type(16))) __bf16 v16bf;
typedef __attribute__((ext_vector_type(8)))  float  v8f;

// 16-bit A-operand K-index pattern for V_WMMA_*_16X16X32 (ISA 7.12.2):
// lanes 0-15: halves 0..7 -> K=0..7,  halves 8..15 -> K=16..23
// lanes 16-31: halves 0..7 -> K=8..15, halves 8..15 -> K=24..31
__device__ __forceinline__ int a_kidx(int lane, int i) {
  int kb = (lane & 16) ? 8 : 0;
  return (i < 8) ? (kb + i) : (16 + kb + (i - 8));
}

__device__ __forceinline__ v8f wmma_bf16(v16bf a, v16bf b, v8f c) {
  return __builtin_amdgcn_wmma_f32_16x16x32_bf16(false, a, false, b, (short)0, c,
                                                 false, false);
}

// reference channel index (c*8+h) from permuted index (h*32+c)
__device__ __forceinline__ int remap_ch(int k) { return ((k & 31) << 3) | (k >> 5); }

// ---------------------------------------------------------------------------
__global__ void k_zero(float* p, int n) {
  for (int i = blockIdx.x * blockDim.x + threadIdx.x; i < n;
       i += gridDim.x * blockDim.x)
    p[i] = 0.f;
}

// LayerNorm (rec_norm applied to both rec and lig, as in reference) -> bf16 X
__global__ void __launch_bounds__(256) k_ln(const float* __restrict__ rec,
                                            const float* __restrict__ lig,
                                            const float* __restrict__ g,
                                            const float* __restrict__ b,
                                            bf16_t* __restrict__ X) {
  int wave = threadIdx.x >> 5, lane = threadIdx.x & 31;
  int row = blockIdx.x * 8 + wave;  // 2944 rows total
  const float* src = (row < 384) ? (rec + (size_t)row * 256)
                                 : (lig + (size_t)(row - 384) * 256);
  float v[8];
  float s = 0.f;
#pragma unroll
  for (int i = 0; i < 8; ++i) { v[i] = src[i * 32 + lane]; s += v[i]; }
#pragma unroll
  for (int m = 1; m < 32; m <<= 1) s += __shfl_xor(s, m, 32);
  float mu = s * (1.f / 256.f);
  float var = 0.f;
#pragma unroll
  for (int i = 0; i < 8; ++i) { float d = v[i] - mu; var += d * d; }
#pragma unroll
  for (int m = 1; m < 32; m <<= 1) var += __shfl_xor(var, m, 32);
  float rstd = rsqrtf(var * (1.f / 256.f) + 1e-5f);
  bf16_t* dst = X + (size_t)row * 256;
#pragma unroll
  for (int i = 0; i < 8; ++i) {
    int c = i * 32 + lane;
    dst[c] = (bf16_t)((v[i] - mu) * rstd * g[c] + b[c]);
  }
}

// Pair-bias weights [128,8] -> bf16 TRANSPOSED [16,128] (heads 8..15 zero),
// scaled by 1/sqrt(32). Layout: wpad[quad*2048 + n*128 + k].
__global__ void k_prep_w(const float* __restrict__ wrr, const float* __restrict__ wrl,
                         const float* __restrict__ wlr, const float* __restrict__ wll,
                         bf16_t* __restrict__ wpad) {
  int i = blockIdx.x * blockDim.x + threadIdx.x;  // 4*16*128 = 8192
  if (i >= 8192) return;
  int quad = i >> 11, n = (i >> 7) & 15, k = i & 127;
  const float* w = (quad == 0) ? wrr : (quad == 1) ? wrl : (quad == 2) ? wlr : wll;
  float v = (n < 8) ? w[k * 8 + n] * 0.17677669529663687f : 0.f;
  wpad[i] = (bf16_t)v;
}

// Pack projection + final weights -> bf16, transposed, permutation folded in.
// mats 0..5 (QKV rq,rk,rv,lq,lk,lv): Wt[w][n*256+k] = W[k*256 + remap_ch(n)]
// mats 6..7 (final rec,lig):         Wt[w][n*256+k] = W[remap_ch(k)*256 + n]
__global__ void k_pack_w(const float* __restrict__ w0, const float* __restrict__ w1,
                         const float* __restrict__ w2, const float* __restrict__ w3,
                         const float* __restrict__ w4, const float* __restrict__ w5,
                         const float* __restrict__ w6, const float* __restrict__ w7,
                         bf16_t* __restrict__ wt) {
  int id = blockIdx.x * blockDim.x + threadIdx.x;  // 8*65536 = 524288
  if (id >= 524288) return;
  int mat = id >> 16, rem = id & 65535;
  int n = rem >> 8, k = rem & 255;
  const float* w = (mat == 0) ? w0 : (mat == 1) ? w1 : (mat == 2) ? w2
                 : (mat == 3) ? w3 : (mat == 4) ? w4 : (mat == 5) ? w5
                 : (mat == 6) ? w6 : w7;
  float v = (mat < 6) ? w[(size_t)k * 256 + remap_ch(n)]
                      : w[(size_t)remap_ch(k) * 256 + n];
  wt[id] = (bf16_t)v;
}

// Fused Q/K/V projection: X[2944,256]bf16 @ Wt -> bf16 [2944,256] permuted.
// One wave computes one 16x16 tile of Q, K and V (A loaded once, 24 WMMAs).
__global__ void __launch_bounds__(256) k_qkv(const bf16_t* __restrict__ X,
                                             const bf16_t* __restrict__ Wt,
                                             bf16_t* __restrict__ Q,
                                             bf16_t* __restrict__ K,
                                             bf16_t* __restrict__ V) {
  int wave = threadIdx.x >> 5, lane = threadIdx.x & 31;
  int task = blockIdx.x * 8 + wave;  // 184*16 = 2944 tasks
  int ct = task & 15, rt = task >> 4;
  int r0 = rt * 16;
  int wbase = (r0 < 384) ? 0 : 3;  // rec vs lig weights; 384=24*16 no straddle
  int n_out = ct * 16 + (lane & 15);
  int mrow = lane & 15, mb = (lane & 16) ? 8 : 0;
  const bf16_t* arow = X + (size_t)(r0 + mrow) * 256;
  const bf16_t* wq = Wt + (size_t)(wbase + 0) * 65536 + (size_t)n_out * 256;
  const bf16_t* wk = Wt + (size_t)(wbase + 1) * 65536 + (size_t)n_out * 256;
  const bf16_t* wv = Wt + (size_t)(wbase + 2) * 65536 + (size_t)n_out * 256;
  v8f cq = {}, ck = {}, cv = {};
#pragma unroll
  for (int k0 = 0; k0 < 256; k0 += 32) {
    v16bf a, bq, bk, bv;
#pragma unroll
    for (int i = 0; i < 16; ++i) {
      int kk = k0 + a_kidx(lane, i);
      a[i]  = arow[kk];
      bq[i] = wq[kk];
      bk[i] = wk[kk];
      bv[i] = wv[kk];
    }
    cq = wmma_bf16(a, bq, cq);
    ck = wmma_bf16(a, bk, ck);
    cv = wmma_bf16(a, bv, cv);
  }
#pragma unroll
  for (int j = 0; j < 8; ++j) {
    size_t o = (size_t)(r0 + mb + j) * 256 + n_out;
    Q[o] = (bf16_t)cq[j];
    K[o] = (bf16_t)ck[j];
    V[o] = (bf16_t)cv[j];
  }
}

// Pair bias: streams pair[1024,1024,128] once (non-temporal: read-once, keep
// L2 for bias/QKV); bias[q,k,h] = (pair@w)*factor via WMMA.
__global__ void __launch_bounds__(256) k_bias(const float* __restrict__ pair,
                                              const bf16_t* __restrict__ wpad,
                                              float* __restrict__ bias) {
  int wave = threadIdx.x >> 5, lane = threadIdx.x & 31;
  int task = blockIdx.x * 8 + wave;  // 1024 q * 64 k-tiles = 65536
  int kt = task & 63, q = task >> 6;
  int quad = ((q >= 384) ? 2 : 0) | ((kt >= 24) ? 1 : 0);  // rr/rl/lr/ll
  int n = lane & 15, mrow = lane & 15, mb = (lane & 16) ? 8 : 0;
  const bf16_t* wrow = wpad + quad * 2048 + n * 128;  // transposed [16][128]
  const float* prow = pair + ((size_t)q * 1024 + kt * 16 + mrow) * 128;
  v8f c = {};
#pragma unroll
  for (int k0 = 0; k0 < 128; k0 += 32) {
    v16bf a, b;
#pragma unroll
    for (int i = 0; i < 16; ++i) {
      int kk = k0 + a_kidx(lane, i);
      a[i] = (bf16_t)__builtin_nontemporal_load(prow + kk);
      b[i] = wrow[kk];
    }
    c = wmma_bf16(a, b, c);
  }
  if (n < 8) {
#pragma unroll
    for (int j = 0; j < 8; ++j)
      bias[((size_t)q * 1024 + kt * 16 + mb + j) * 8 + n] = c[j];
  }
}

// Flash attention: one wave = (m, head, 16-query tile); the 8 waves of a
// block share (m, head) over consecutive query tiles so K/V reads hit cache.
// 32 keys per iteration: 2 QK WMMAs + online softmax + 2 AV WMMAs.
__global__ void __launch_bounds__(256) k_attn(const bf16_t* __restrict__ Q,
                                              const bf16_t* __restrict__ K,
                                              const bf16_t* __restrict__ V,
                                              const float* __restrict__ bias,
                                              float* __restrict__ att) {
  __shared__ float Pst[8][16 * 32];
  int wave = threadIdx.x >> 5, lane = threadIdx.x & 31;
  int task = blockIdx.x * 8 + wave;  // 4*8*64 = 2048
  int qt = task & 63;                // low bits: query tile (block shares m,h)
  int mh = task >> 6;
  int h = mh & 7, m = mh >> 3;
  int qbase = qt * 16;
  bool qrec = (qbase < 384);
  int n = lane & 15, mrow = lane & 15, mb = (lane & 16) ? 8 : 0;
  float* Pw = &Pst[wave][0];

  // A operand: Q tile (16 queries x 32 head channels)
  int qg = qbase + mrow;
  int qr = (qg < 384) ? qg : (384 + m * 640 + (qg - 384));
  const bf16_t* qp = Q + (size_t)qr * 256 + h * 32;
  v16bf aq;
#pragma unroll
  for (int i = 0; i < 16; ++i) aq[i] = qp[a_kidx(lane, i)];

  v8f accR0 = {}, accR1 = {}, accL0 = {}, accL1 = {};  // rec/lig keys x c-half
  float rmax[8], rsum[8];
#pragma unroll
  for (int j = 0; j < 8; ++j) { rmax[j] = -1e30f; rsum[j] = 0.f; }

  for (int kt2 = 0; kt2 < 32; ++kt2) {
    int kbase = kt2 * 32;  // 384 = 12*32: key groups never straddle rec/lig
    v8f c0 = {}, c1 = {};
#pragma unroll
    for (int s = 0; s < 2; ++s) {
      int key = kbase + s * 16 + n;
      int kr = (key < 384) ? key : (384 + m * 640 + (key - 384));
      const bf16_t* kp = K + (size_t)kr * 256 + h * 32;
      v16bf bk;
#pragma unroll
      for (int i = 0; i < 16; ++i) bk[i] = kp[a_kidx(lane, i)];
      v8f cc = {};
      cc = wmma_bf16(aq, bk, cc);
      const float* bp = bias + ((size_t)(qbase + mb) * 1024 + key) * 8 + h;
#pragma unroll
      for (int j = 0; j < 8; ++j) cc[j] += bp[(size_t)j * 8192];
      if (s == 0) c0 = cc; else c1 = cc;
    }
    // online softmax update (row stats replicated across each 16-lane half)
    float p0[8], p1[8];
#pragma unroll
    for (int j = 0; j < 8; ++j) {
      float t = fmaxf(c0[j], c1[j]);
#pragma unroll
      for (int msk = 1; msk < 16; msk <<= 1) t = fmaxf(t, __shfl_xor(t, msk, 32));
      float nm = fmaxf(rmax[j], t);
      float sc = __expf(rmax[j] - nm);
      rmax[j] = nm;
      p0[j] = __expf(c0[j] - nm);
      p1[j] = __expf(c1[j] - nm);
      float ts = p0[j] + p1[j];
#pragma unroll
      for (int msk = 1; msk < 16; msk <<= 1) ts += __shfl_xor(ts, msk, 32);
      rsum[j] = rsum[j] * sc + ts;
      accR0[j] *= sc; accR1[j] *= sc; accL0[j] *= sc; accL1[j] *= sc;
    }
    // transpose P (C-layout) -> A-layout via per-wave LDS
#pragma unroll
    for (int j = 0; j < 8; ++j) {
      Pw[(mb + j) * 32 + n] = p0[j];
      Pw[(mb + j) * 32 + 16 + n] = p1[j];
    }
    asm volatile("s_wait_dscnt 0" ::: "memory");
    v16bf ap;
#pragma unroll
    for (int i = 0; i < 16; ++i) ap[i] = (bf16_t)Pw[mrow * 32 + a_kidx(lane, i)];

    bool recKeys = (kbase < 384);  // wave-uniform
#pragma unroll
    for (int ch = 0; ch < 2; ++ch) {
      v16bf bv;
#pragma unroll
      for (int i = 0; i < 16; ++i) {
        int key = kbase + a_kidx(lane, i);
        int vr = (key < 384) ? key : (384 + m * 640 + (key - 384));
        bv[i] = V[(size_t)vr * 256 + h * 32 + ch * 16 + n];
      }
      if (recKeys) {
        if (ch == 0) accR0 = wmma_bf16(ap, bv, accR0);
        else         accR1 = wmma_bf16(ap, bv, accR1);
      } else {
        if (ch == 0) accL0 = wmma_bf16(ap, bv, accL0);
        else         accL1 = wmma_bf16(ap, bv, accL1);
      }
    }
  }

  float inv[8];
#pragma unroll
  for (int j = 0; j < 8; ++j) inv[j] = 1.f / rsum[j];
  if (qrec) {
    // rec_out: m==0's rec-key part + 0.25 * mean-over-m lig-key part
#pragma unroll
    for (int j = 0; j < 8; ++j) {
      int qg2 = qbase + mb + j;
      float v0 = 0.25f * accL0[j] * inv[j] + ((m == 0) ? accR0[j] * inv[j] : 0.f);
      float v1 = 0.25f * accL1[j] * inv[j] + ((m == 0) ? accR1[j] * inv[j] : 0.f);
      atomicAdd(&att[(size_t)qg2 * 256 + h * 32 + n], v0);
      atomicAdd(&att[(size_t)qg2 * 256 + h * 32 + 16 + n], v1);
    }
  } else {
#pragma unroll
    for (int j = 0; j < 8; ++j) {
      int row = 384 + m * 640 + (qbase - 384) + mb + j;
      att[(size_t)row * 256 + h * 32 + n] = (accR0[j] + accL0[j]) * inv[j];
      att[(size_t)row * 256 + h * 32 + 16 + n] = (accR1[j] + accL1[j]) * inv[j];
    }
  }
}

// Final projection: att[2944,256] @ w_final + b -> d_out (rec then lig, f32)
__global__ void __launch_bounds__(256) k_final(const float* __restrict__ att,
                                               const bf16_t* __restrict__ Wt,
                                               const float* __restrict__ br,
                                               const float* __restrict__ bl,
                                               float* __restrict__ out) {
  int wave = threadIdx.x >> 5, lane = threadIdx.x & 31;
  int task = blockIdx.x * 8 + wave;  // 184*16 = 2944
  int ct = task & 15, rt = task >> 4;
  int r0 = rt * 16;
  bool isrec = (r0 < 384);
  int n_out = ct * 16 + (lane & 15);
  int mrow = lane & 15, mb = (lane & 16) ? 8 : 0;
  const float* arow = att + (size_t)(r0 + mrow) * 256;
  const bf16_t* wrow = Wt + (size_t)(isrec ? 6 : 7) * 65536 + (size_t)n_out * 256;
  v8f c = {};
#pragma unroll
  for (int k0 = 0; k0 < 256; k0 += 32) {
    v16bf a, b;
#pragma unroll
    for (int i = 0; i < 16; ++i) {
      int kk = k0 + a_kidx(lane, i);
      a[i] = (bf16_t)arow[kk];
      b[i] = wrow[kk];
    }
    c = wmma_bf16(a, b, c);
  }
  float bias = (isrec ? br : bl)[n_out];
#pragma unroll
  for (int j = 0; j < 8; ++j) {
    int row = r0 + mb + j;
    float v = c[j] + bias;
    if (row < 384) out[(size_t)row * 256 + n_out] = v;
    else out[98304 + (size_t)(row - 384) * 256 + n_out] = v;
  }
}

// ---------------------------------------------------------------------------
extern "C" void kernel_launch(void* const* d_in, const int* in_sizes, int n_in,
                              void* d_out, int out_size, void* d_ws, size_t ws_size,
                              hipStream_t stream) {
  const float* rec  = (const float*)d_in[0];
  const float* lig  = (const float*)d_in[1];
  const float* pair = (const float*)d_in[2];
  const float* ln_g = (const float*)d_in[3];
  const float* ln_b = (const float*)d_in[4];
  const float* wrq  = (const float*)d_in[5];
  const float* wrk  = (const float*)d_in[6];
  const float* wrv  = (const float*)d_in[7];
  const float* wlq  = (const float*)d_in[8];
  const float* wlk  = (const float*)d_in[9];
  const float* wlv  = (const float*)d_in[10];
  const float* wrr  = (const float*)d_in[11];
  const float* wll  = (const float*)d_in[12];
  const float* wrl  = (const float*)d_in[13];
  const float* wlr  = (const float*)d_in[14];
  const float* wrf  = (const float*)d_in[15];
  const float* brf  = (const float*)d_in[16];
  const float* wlf  = (const float*)d_in[17];
  const float* blf  = (const float*)d_in[18];
  float* out = (float*)d_out;

  char* ws = (char*)d_ws;
  bf16_t* X    = (bf16_t*)(ws + 0);          // 2944*256*2  = 1507328 B
  bf16_t* Q    = (bf16_t*)(ws + 1507328);
  bf16_t* K    = (bf16_t*)(ws + 3014656);
  bf16_t* V    = (bf16_t*)(ws + 4521984);
  bf16_t* Wpad = (bf16_t*)(ws + 6029312);    // 4*16*128*2  = 16384 B
  bf16_t* Wt   = (bf16_t*)(ws + 6045696);    // 8*256*256*2 = 1048576 B
  float*  att  = (float*) (ws + 7094272);    // 2944*256*4  = 3014656 B
  float*  bias = (float*) (ws + 10108928);   // 1024*1024*8*4 = 33554432 B
                                             // total 43663360 B (~41.6 MiB)

  k_zero  <<<736, 256, 0, stream>>>(att, 2944 * 256);
  k_ln    <<<368, 256, 0, stream>>>(rec, lig, ln_g, ln_b, X);
  k_prep_w<<<32, 256, 0, stream>>>(wrr, wrl, wlr, wll, Wpad);
  k_pack_w<<<2048, 256, 0, stream>>>(wrq, wrk, wrv, wlq, wlk, wlv, wrf, wlf, Wt);
  k_qkv   <<<368, 256, 0, stream>>>(X, Wt, Q, K, V);
  k_bias  <<<8192, 256, 0, stream>>>(pair, Wpad, bias);
  k_attn  <<<256, 256, 0, stream>>>(Q, K, V, bias, att);
  k_final <<<368, 256, 0, stream>>>(att, Wt, brf, blf, out);
}
